// LLaMALayer_47218870452984
// MI455X (gfx1250) — matrix operand order repaired and checked by reading, not compiled, not run on hardware
//
#include <hip/hip_runtime.h>

typedef unsigned short ushort_t;
typedef unsigned int   uint_t;

typedef __attribute__((ext_vector_type(16))) __bf16 v16bf;
typedef __attribute__((ext_vector_type(8)))  float  v8f;

union Frag {
    v16bf  v;
    uint_t d[8];
};

__device__ __forceinline__ ushort_t f2bf(float f) {
    uint_t u = __float_as_uint(f);
    u += 0x7FFFu + ((u >> 16) & 1u);   // round-to-nearest-even
    return (ushort_t)(u >> 16);
}

// ---------------------------------------------------------------------------
// CDNA5 async global->LDS copy (16B per lane), tracked by ASYNCcnt.
// Generic LDS pointer low 32 bits == wave-relative LDS byte address.
// ---------------------------------------------------------------------------
__device__ __forceinline__ void async_ld16(const ushort_t* g, ushort_t* l) {
    unsigned long long ga = (unsigned long long)(size_t)g;
    unsigned la           = (unsigned)(size_t)l;
    asm volatile("global_load_async_to_lds_b128 %0, %1, off"
                 :: "v"(la), "v"(ga) : "memory");
}
#define ASYNC_WAIT_0() asm volatile("s_wait_asynccnt 0x0" ::: "memory")
#define ASYNC_WAIT_4() asm volatile("s_wait_asynccnt 0x4" ::: "memory")

// ---------------------------------------------------------------------------
// Model constants
// ---------------------------------------------------------------------------
#define BATCH   4
#define SEQ     1024
#define BS      (BATCH * SEQ)   // 4096 rows
#define DM      2048            // d_model
#define NH      16
#define DH      128
#define HKV     4               // kv heads
#define DKV     (HKV * DH)      // 512
#define DFF     2048

// ---------------------------------------------------------------------------
// fp32 [K][N] -> bf16 transposed [N][K]   (weights; makes GEMM B-tiles linear)
// ---------------------------------------------------------------------------
__global__ void __launch_bounds__(256)
cvt_transpose_bf16(const float* __restrict__ W, ushort_t* __restrict__ WT,
                   int K, int N) {
    size_t total = (size_t)K * N;
    for (size_t o = (size_t)blockIdx.x * 256 + threadIdx.x; o < total;
         o += (size_t)gridDim.x * 256) {
        size_t n = o / K;
        size_t k = o - n * K;
        WT[o] = f2bf(W[k * N + n]);
    }
}

// ---------------------------------------------------------------------------
// bf16 V [BS][DKV] -> bf16 V^T [(b*HKV+g)*DH + d][SEQ]
// ---------------------------------------------------------------------------
__global__ void __launch_bounds__(256)
transpose_v_bf16(const ushort_t* __restrict__ V, ushort_t* __restrict__ VT) {
    size_t total = (size_t)BS * DKV;
    for (size_t o = (size_t)blockIdx.x * 256 + threadIdx.x; o < total;
         o += (size_t)gridDim.x * 256) {
        int s = (int)(o % SEQ);
        size_t t = o / SEQ;
        int d = (int)(t % DH);
        size_t t2 = t / DH;
        int g = (int)(t2 % HKV);
        int b = (int)(t2 / HKV);
        VT[o] = V[(size_t)(b * SEQ + s) * DKV + g * DH + d];
    }
}

// ---------------------------------------------------------------------------
// RMSNorm: one row (2048 elems) per block, output bf16
// ---------------------------------------------------------------------------
__global__ void __launch_bounds__(256)
rmsnorm_bf16(const float* __restrict__ x, const float* __restrict__ scale,
             ushort_t* __restrict__ xn) {
    __shared__ float red[8];
    const int row  = blockIdx.x;
    const int tid  = threadIdx.x;
    const int lane = tid & 31;
    const int wv   = tid >> 5;
    const float* xr = x + (size_t)row * DM;

    float ss = 0.f;
#pragma unroll
    for (int i = 0; i < 8; ++i) {
        float v = xr[tid + i * 256];
        ss += v * v;
    }
#pragma unroll
    for (int off = 16; off > 0; off >>= 1) ss += __shfl_xor(ss, off, 32);
    if (lane == 0) red[wv] = ss;
    __syncthreads();
    float tot = 0.f;
#pragma unroll
    for (int w = 0; w < 8; ++w) tot += red[w];
    const float inv = rsqrtf(tot / (float)DM + 1e-6f);
#pragma unroll
    for (int i = 0; i < 8; ++i) {
        int c = tid + i * 256;
        xn[(size_t)row * DM + c] = f2bf(scale[c] * xr[c] * inv);
    }
}

// ---------------------------------------------------------------------------
// WMMA bf16 GEMM:  C[M,N] = A[M,K] * Bt[N,K]^T + bias  (+ epilogue)
//   EPI 0: store bf16
//   EPI 1: h = val; store bf16( h * sigmoid(h) * h )
//   EPI 2: store f32( resid + val )
// Block 256 thr = 8 waves.  Tile BM=128 x BN=128, BK=32, double-buffered LDS
// staged with global_load_async_to_lds_b128.  Wave (wm 0..3, wn 0..1)
// computes 32x64 = 2x4 WMMA tiles (A-frags reused across 4 B-frags).
// ---------------------------------------------------------------------------
#define GPITCH 40   // LDS pitch in bf16 elems (80B: 16B-aligned rows, no pow2)

template <int EPI>
__global__ void __launch_bounds__(256)
gemm_bf16(const ushort_t* __restrict__ A, const ushort_t* __restrict__ Bt,
          const float* __restrict__ bias, const float* __restrict__ resid,
          ushort_t* __restrict__ outb, float* __restrict__ outf,
          int M, int N, int K) {
    __shared__ __align__(16) ushort_t As[2][128 * GPITCH];   // [m][k]
    __shared__ __align__(16) ushort_t Bs[2][128 * GPITCH];   // [n][k]

    const int tid  = threadIdx.x;
    const int lane = tid & 31;
    const int wv   = tid >> 5;
    const int wm   = wv >> 1;       // 0..3
    const int wn   = wv & 1;        // 0..1
    const int half = lane >> 4;
    const int l16  = lane & 15;
    const int m0   = blockIdx.x * 128;
    const int n0   = blockIdx.y * 128;

    v8f acc[2][4];
#pragma unroll
    for (int i = 0; i < 2; ++i)
#pragma unroll
        for (int j = 0; j < 4; ++j) acc[i][j] = (v8f)0.f;

    // stage one 128x32 A tile + 128x32 B tile into buffer `buf` (async)
    auto stage = [&](int buf, int k0) {
#pragma unroll
        for (int i = 0; i < 2; ++i) {
            int c = tid + i * 256;            // 512 16B-chunks
            int row = c >> 2, q = c & 3;
            async_ld16(A + (size_t)(m0 + row) * K + k0 + q * 8,
                       &As[buf][row * GPITCH + q * 8]);
        }
#pragma unroll
        for (int i = 0; i < 2; ++i) {
            int c = tid + i * 256;
            int row = c >> 2, q = c & 3;
            async_ld16(Bt + (size_t)(n0 + row) * K + k0 + q * 8,
                       &Bs[buf][row * GPITCH + q * 8]);
        }
    };

    stage(0, 0);
    const int NK = K >> 5;
    for (int it = 0; it < NK; ++it) {
        const int cur = it & 1;
        if (it + 1 < NK) {
            stage(cur ^ 1, (it + 1) * 32);
            ASYNC_WAIT_4();      // previous stage (4 ops/thread) complete
        } else {
            ASYNC_WAIT_0();
        }
        __syncthreads();         // all waves staged `cur` + done reading it-1

        const uint_t* Au = (const uint_t*)As[cur];
        const uint_t* Bu = (const uint_t*)Bs[cur];
        Frag af[2], bf[4];
#pragma unroll
        for (int mi = 0; mi < 2; ++mi) {
            int row = wm * 32 + mi * 16 + l16;
#pragma unroll
            for (int r = 0; r < 8; ++r)
                af[mi].d[r] = Au[row * (GPITCH / 2) + 4 * half + (r & 3) + 8 * (r >> 2)];
        }
#pragma unroll
        for (int ni = 0; ni < 4; ++ni) {
            int n = wn * 64 + ni * 16 + l16;
#pragma unroll
            for (int r = 0; r < 8; ++r)
                bf[ni].d[r] = Bu[n * (GPITCH / 2) + 4 * half + (r & 3) + 8 * (r >> 2)];
        }
#pragma unroll
        for (int mi = 0; mi < 2; ++mi)
#pragma unroll
            for (int ni = 0; ni < 4; ++ni)
                acc[mi][ni] = __builtin_amdgcn_wmma_f32_16x16x32_bf16(
                    false, af[mi].v, false, bf[ni].v,
                    (short)0, acc[mi][ni], false, false);
        __syncthreads();         // done reading `cur` before it is restaged
    }

    // ---- epilogue ----
#pragma unroll
    for (int mi = 0; mi < 2; ++mi)
#pragma unroll
        for (int ni = 0; ni < 4; ++ni)
#pragma unroll
            for (int r = 0; r < 8; ++r) {
                int row = m0 + wm * 32 + mi * 16 + half * 8 + r;
                int col = n0 + wn * 64 + ni * 16 + l16;
                float val = acc[mi][ni][r] + bias[col];
                size_t idx = (size_t)row * N + col;
                if (EPI == 0) {
                    outb[idx] = f2bf(val);
                } else if (EPI == 1) {
                    float sg = 1.f / (1.f + __expf(-val));
                    outb[idx] = f2bf(val * sg * val);
                } else {
                    outf[idx] = resid[idx] + val;
                }
            }
}

// ---------------------------------------------------------------------------
// Flash attention (GQA).  Grid: (B*NH, SEQ/128).  256 threads = 8 waves.
// K/V chunks (32 keys) double-buffered in LDS, staged asynchronously so the
// next chunk streams in while WMMAs+softmax consume the current one.
// Q: bf16 [BS][2048];  K: bf16 [BS][512];  Vt: bf16 [(b*HKV+g)*DH+d][SEQ]
// ---------------------------------------------------------------------------
#define KPITCH 136   // 272B rows (16B aligned)
#define VPITCH 40    // 80B rows

__global__ void __launch_bounds__(256)
attn_gqa(const ushort_t* __restrict__ Q, const ushort_t* __restrict__ Kb,
         const ushort_t* __restrict__ VtG, ushort_t* __restrict__ O) {
    __shared__ __align__(16) ushort_t Kl[2][32 * KPITCH];   // [key][d]
    __shared__ __align__(16) ushort_t Vt[2][128 * VPITCH];  // [d][key]
    __shared__ __align__(16) ushort_t Pl[8 * 16 * 32];      // per-wave P tiles

    const int tid  = threadIdx.x;
    const int lane = tid & 31;
    const int wv   = tid >> 5;
    const int half = lane >> 4;
    const int l16  = lane & 15;

    const int bh  = blockIdx.x;
    const int b   = bh >> 4;
    const int h   = bh & 15;
    const int hkv = h >> 2;
    const int qbase = blockIdx.y * 128 + wv * 16;

    // ---- load Q fragments (16 queries x 128 dims = 4 A-frags) ----
    Frag qf[4];
    {
        const uint_t* Qd = (const uint_t*)Q;
        size_t qoff = (size_t)(b * SEQ + qbase + l16) * (DM / 2) + (h * DH) / 2;
#pragma unroll
        for (int dc = 0; dc < 4; ++dc)
#pragma unroll
            for (int r = 0; r < 8; ++r)
                qf[dc].d[r] = Qd[qoff + dc * 16 + 4 * half + (r & 3) + 8 * (r >> 2)];
    }

    v8f acc_o[8];
#pragma unroll
    for (int t = 0; t < 8; ++t) acc_o[t] = (v8f)0.f;
    float mrun[8], lrun[8];
#pragma unroll
    for (int r = 0; r < 8; ++r) { mrun[r] = -1e30f; lrun[r] = 0.f; }

    const float sc = 0.08838834764831845f;   // 1/sqrt(128)
    const size_t vbase = (size_t)(b * HKV + hkv) * DH * SEQ;

    // async stage of one 32-key K chunk + V^T chunk into buffer `buf`
    auto stage = [&](int buf, int kc) {
#pragma unroll
        for (int i = 0; i < 2; ++i) {
            int c = tid + i * 256;            // 512 chunks of 16B
            int key = c >> 4, q = c & 15;
            async_ld16(Kb + (size_t)(b * SEQ + kc * 32 + key) * DKV + hkv * DH + q * 8,
                       &Kl[buf][key * KPITCH + q * 8]);
        }
#pragma unroll
        for (int i = 0; i < 2; ++i) {
            int c = tid + i * 256;
            int d_ = c >> 2, q = c & 3;
            async_ld16(VtG + vbase + (size_t)d_ * SEQ + kc * 32 + q * 8,
                       &Vt[buf][d_ * VPITCH + q * 8]);
        }
    };

    stage(0, 0);
    const int NC = SEQ / 32;
    for (int kc = 0; kc < NC; ++kc) {
        const int cur = kc & 1;
        if (kc + 1 < NC) {
            stage(cur ^ 1, kc + 1);
            ASYNC_WAIT_4();      // previous stage (4 ops/thread) complete
        } else {
            ASYNC_WAIT_0();
        }
        __syncthreads();

        const uint_t* Ku = (const uint_t*)Kl[cur];
        const uint_t* Vu = (const uint_t*)Vt[cur];

        // ---- S = Q * K^T  (two 16x16 tiles over 32 keys) ----
        v8f s[2];
#pragma unroll
        for (int nt = 0; nt < 2; ++nt) {
            s[nt] = (v8f)0.f;
            int key = nt * 16 + l16;
#pragma unroll
            for (int dc = 0; dc < 4; ++dc) {
                Frag bfr;
#pragma unroll
                for (int r = 0; r < 8; ++r)
                    bfr.d[r] = Ku[key * (KPITCH / 2) + dc * 16 + 4 * half + (r & 3) + 8 * (r >> 2)];
                s[nt] = __builtin_amdgcn_wmma_f32_16x16x32_bf16(
                    false, qf[dc].v, false, bfr.v, (short)0, s[nt], false, false);
            }
        }

        // ---- online softmax ----
        float fac[8];
#pragma unroll
        for (int r = 0; r < 8; ++r) {
            float a0 = s[0][r] * sc;
            float a1 = s[1][r] * sc;
            float mx = fmaxf(a0, a1);
#pragma unroll
            for (int off = 1; off < 16; off <<= 1)
                mx = fmaxf(mx, __shfl_xor(mx, off, 32));
            float newm = fmaxf(mrun[r], mx);
            float f    = __expf(mrun[r] - newm);
            float p0   = __expf(a0 - newm);
            float p1   = __expf(a1 - newm);
            s[0][r] = p0;
            s[1][r] = p1;
            float su = p0 + p1;
#pragma unroll
            for (int off = 1; off < 16; off <<= 1)
                su += __shfl_xor(su, off, 32);
            lrun[r] = lrun[r] * f + su;
            mrun[r] = newm;
            fac[r]  = f;
        }
#pragma unroll
        for (int t = 0; t < 8; ++t)
#pragma unroll
            for (int r = 0; r < 8; ++r) acc_o[t][r] *= fac[r];

        // ---- write P to per-wave LDS tile (C-layout -> row-major) ----
        ushort_t* Pw = Pl + wv * 512;
#pragma unroll
        for (int nt = 0; nt < 2; ++nt)
#pragma unroll
            for (int r = 0; r < 8; ++r)
                Pw[(r + 8 * half) * 32 + nt * 16 + l16] = f2bf(s[nt][r]);
        asm volatile("s_wait_dscnt 0" ::: "memory");

        // ---- O += P * V  (K=32 keys, 8 d-tiles) ----
        Frag pa;
        const uint_t* Pu = (const uint_t*)(Pl + wv * 512);
#pragma unroll
        for (int r = 0; r < 8; ++r)
            pa.d[r] = Pu[l16 * 16 + 4 * half + (r & 3) + 8 * (r >> 2)];
#pragma unroll
        for (int t = 0; t < 8; ++t) {
            Frag vb;
#pragma unroll
            for (int r = 0; r < 8; ++r)
                vb.d[r] = Vu[(t * 16 + l16) * (VPITCH / 2) + 4 * half + (r & 3) + 8 * (r >> 2)];
            acc_o[t] = __builtin_amdgcn_wmma_f32_16x16x32_bf16(
                false, pa.v, false, vb.v, (short)0, acc_o[t], false, false);
        }
        __syncthreads();         // done reading `cur` before it is restaged
    }

    // ---- normalize and store ----
#pragma unroll
    for (int t = 0; t < 8; ++t)
#pragma unroll
        for (int r = 0; r < 8; ++r) {
            int rowq = qbase + 8 * half + r;
            O[(size_t)(b * SEQ + rowq) * DM + h * DH + t * 16 + l16] =
                f2bf(acc_o[t][r] / lrun[r]);
        }
}

// ---------------------------------------------------------------------------
// Host side
// ---------------------------------------------------------------------------
extern "C" void kernel_launch(void* const* d_in, const int* in_sizes, int n_in,
                              void* d_out, int out_size, void* d_ws, size_t ws_size,
                              hipStream_t stream) {
    const float* x   = (const float*)d_in[0];
    const float* sc1 = (const float*)d_in[1];
    const float* Wq  = (const float*)d_in[2];
    const float* bq  = (const float*)d_in[3];
    const float* Wk  = (const float*)d_in[4];
    const float* bk  = (const float*)d_in[5];
    const float* Wv  = (const float*)d_in[6];
    const float* bv  = (const float*)d_in[7];
    const float* Wo  = (const float*)d_in[8];
    const float* bo  = (const float*)d_in[9];
    const float* W1  = (const float*)d_in[10];
    const float* b1  = (const float*)d_in[11];
    const float* W2  = (const float*)d_in[12];
    const float* b2  = (const float*)d_in[13];
    float* out = (float*)d_out;

    unsigned char* ws = (unsigned char*)d_ws;
    size_t off = 0;
    auto alloc = [&](size_t elems) -> ushort_t* {
        off = (off + 255) & ~(size_t)255;
        ushort_t* p = (ushort_t*)(ws + off);
        off += elems * sizeof(ushort_t);
        return p;
    };
    ushort_t* xnB = alloc((size_t)BS * DM);
    ushort_t* wqT = alloc((size_t)DM * DM);     // [N][K]
    ushort_t* wkT = alloc((size_t)DKV * DM);
    ushort_t* wvT = alloc((size_t)DKV * DM);
    ushort_t* woT = alloc((size_t)DM * DM);
    ushort_t* w1T = alloc((size_t)DFF * DM);
    ushort_t* w2T = alloc((size_t)DM * DFF);
    ushort_t* qB  = alloc((size_t)BS * DM);
    ushort_t* kB  = alloc((size_t)BS * DKV);
    ushort_t* vB  = alloc((size_t)BS * DKV);
    ushort_t* vtB = alloc((size_t)BS * DKV);
    ushort_t* aoB = alloc((size_t)BS * DM);
    ushort_t* hB  = alloc((size_t)BS * DFF);

    // 1. rmsnorm -> bf16 xn
    rmsnorm_bf16<<<BS, 256, 0, stream>>>(x, sc1, xnB);

    // 2. weight convert + transpose (bf16, [N][K])
    auto cvtT = [&](const float* src, ushort_t* dst, int K, int N) {
        int n = K * N;
        int blocks = (n + 255) / 256;
        if (blocks > 8192) blocks = 8192;
        cvt_transpose_bf16<<<blocks, 256, 0, stream>>>(src, dst, K, N);
    };
    cvtT(Wq, wqT, DM, DM);
    cvtT(Wk, wkT, DM, DKV);
    cvtT(Wv, wvT, DM, DKV);
    cvtT(Wo, woT, DM, DM);
    cvtT(W1, w1T, DM, DFF);
    cvtT(W2, w2T, DFF, DM);

    // 3. Q/K/V projections
    gemm_bf16<0><<<dim3(BS / 128, DM / 128), 256, 0, stream>>>(
        xnB, wqT, bq, nullptr, qB, nullptr, BS, DM, DM);
    gemm_bf16<0><<<dim3(BS / 128, DKV / 128), 256, 0, stream>>>(
        xnB, wkT, bk, nullptr, kB, nullptr, BS, DKV, DM);
    gemm_bf16<0><<<dim3(BS / 128, DKV / 128), 256, 0, stream>>>(
        xnB, wvT, bv, nullptr, vB, nullptr, BS, DKV, DM);

    // 4. V transpose for attention staging
    transpose_v_bf16<<<4096, 256, 0, stream>>>(vB, vtB);

    // 5. attention
    attn_gqa<<<dim3(BATCH * NH, SEQ / 128), 256, 0, stream>>>(qB, kB, vtB, aoB);

    // 6. O projection + residual -> d_out (f32)
    gemm_bf16<2><<<dim3(BS / 128, DM / 128), 256, 0, stream>>>(
        aoB, woT, bo, x, nullptr, out, BS, DM, DM);

    // 7. FFN up + h*silu(h) -> bf16 h   (consumes xn, matching reference bug)
    gemm_bf16<1><<<dim3(BS / 128, DFF / 128), 256, 0, stream>>>(
        xnB, w1T, b1, nullptr, hB, nullptr, BS, DFF, DM);

    // 8. FFN down + residual -> d_out (f32)
    gemm_bf16<2><<<dim3(BS / 128, DM / 128), 256, 0, stream>>>(
        hB, w2T, b2, out, nullptr, out, BS, DM, DFF);
}
